// Multi_Head_Self_Attention_70411693851048
// MI455X (gfx1250) — matrix operand order repaired and checked by reading, not compiled, not run on hardware
//
#include <hip/hip_runtime.h>
#include <hip/hip_bf16.h>

// MHSA for B=2,S=2048,H=2048,NH=16,HD=128 on gfx1250 (wave32, WMMA).
// Outputs: [x_out f32 (B*S*H)] ++ [attn (B*NH*S*S)], concatenated in d_out.
// V projection is dead code in the reference -> only Q,K are computed.

typedef __attribute__((ext_vector_type(16))) _Float16 v16h;
typedef __attribute__((ext_vector_type(8)))  float    v8f;

#define Bc 2
#define Sc 2048
#define Hc 2048
#define NHc 16
#define HDc 128
#define MROWS (Bc * Sc)     // 4096
#define QKN (2 * Hc)        // 4096 (q|k concatenated columns)

#define WMMA_F16(a, b, c) \
  __builtin_amdgcn_wmma_f32_16x16x32_f16(false, (a), false, (b), (short)0, (c), false, false)

// ---- fragment loaders (ISA 7.12.2 layouts) -------------------------------
// A 16x32 f16: lane L holds row M=(L&15); K chunks [(L>>4)*8 .. +7] and +16.
__device__ __forceinline__ v16h load_a_frag(const _Float16* __restrict__ p) {
  union { v16h v; uint4 u[2]; } t;
  t.u[0] = *(const uint4*)(p);
  t.u[1] = *(const uint4*)(p + 16);
  return t.v;
}
// B 32x16 f16: lane L holds col N=(L&15); K chunks [(L>>4)*8 .. +7] and +16.
__device__ __forceinline__ v16h load_b_frag(const _Float16* __restrict__ p) {
  union { v16h v; uint4 u[2]; } t;
  t.u[0] = *(const uint4*)(p);
  t.u[1] = *(const uint4*)(p + 16);
  return t.v;
}

// ---- f32 -> f16 convert, vectorized (float4 in, 8B packed f16 out) -------
__global__ void mhsa_cvt_f16(const float* __restrict__ src,
                             _Float16* __restrict__ dst, size_t n4) {
  size_t i = (size_t)blockIdx.x * blockDim.x + threadIdx.x;
  size_t stride = (size_t)gridDim.x * blockDim.x;
  for (; i < n4; i += stride) {
    float4 f = *(const float4*)(src + i * 4);
    union { _Float16 h[4]; uint2 u; } t;
    t.h[0] = (_Float16)f.x; t.h[1] = (_Float16)f.y;
    t.h[2] = (_Float16)f.z; t.h[3] = (_Float16)f.w;
    *(uint2*)(dst + i * 4) = t.u;
  }
}

// ---- generic C = A * Bw^T + bias, WMMA f16 -> f32 accum -------------------
// A: [M,K] row-major f16.  Bw: [N,K] row-major f16 (K contiguous for both).
// Wave tile 32x64 (2 A-frags x 4 B-frags -> 8 independent accumulators).
// Software-pipelined: next k-step's fragments are in flight during WMMAs.
// Block: 8 waves as 2(m) x 4(n) -> 64 x 256 block tile. Grid: (N/256, M/64).
template <bool F16OUT>
__global__ void mhsa_gemm_wmma(const _Float16* __restrict__ A,
                               const _Float16* __restrict__ Bw,
                               const float* __restrict__ bias,
                               void* __restrict__ Cout,
                               int K, int ldc) {
  const int lane = threadIdx.x & 31;
  const int wave = threadIdx.x >> 5;
  const int mw = wave & 1, nw = wave >> 1;
  const int m0 = blockIdx.y * 64 + mw * 32;
  const int n0 = blockIdx.x * 256 + nw * 64;

  const int aoff = (lane >> 4) << 3;   // A: K chunk base (0 or 8)
  const int boff = (lane >> 4) << 3;   // B: K chunk base (0 or 8)
  const _Float16* pa = A  + (size_t)(m0 + (lane & 15)) * K + aoff;
  const _Float16* pb = Bw + (size_t)(n0 + (lane & 15)) * K + boff;
  const size_t K16 = (size_t)16 * K;

  v8f c[2][4] = {};

  // prologue: fragments for k = 0
  v16h a0 = load_a_frag(pa);
  v16h a1 = load_a_frag(pa + K16);
  v16h b0 = load_b_frag(pb);
  v16h b1 = load_b_frag(pb + K16);
  v16h b2 = load_b_frag(pb + 2 * K16);
  v16h b3 = load_b_frag(pb + 3 * K16);

#pragma unroll 2
  for (int k = 32; k < K; k += 32) {
    // issue next k-step's loads before this step's WMMAs
    v16h na0 = load_a_frag(pa + k);
    v16h na1 = load_a_frag(pa + K16 + k);
    v16h nb0 = load_b_frag(pb + k);
    v16h nb1 = load_b_frag(pb + K16 + k);
    v16h nb2 = load_b_frag(pb + 2 * K16 + k);
    v16h nb3 = load_b_frag(pb + 3 * K16 + k);
    __builtin_prefetch(pa + k + 512, 0, 1);   // global_prefetch_b8
    __builtin_prefetch(pb + k + 512, 0, 1);
    c[0][0] = WMMA_F16(a0, b0, c[0][0]);
    c[0][1] = WMMA_F16(a0, b1, c[0][1]);
    c[0][2] = WMMA_F16(a0, b2, c[0][2]);
    c[0][3] = WMMA_F16(a0, b3, c[0][3]);
    c[1][0] = WMMA_F16(a1, b0, c[1][0]);
    c[1][1] = WMMA_F16(a1, b1, c[1][1]);
    c[1][2] = WMMA_F16(a1, b2, c[1][2]);
    c[1][3] = WMMA_F16(a1, b3, c[1][3]);
    a0 = na0; a1 = na1; b0 = nb0; b1 = nb1; b2 = nb2; b3 = nb3;
  }
  // epilogue: last k-step
  c[0][0] = WMMA_F16(a0, b0, c[0][0]);
  c[0][1] = WMMA_F16(a0, b1, c[0][1]);
  c[0][2] = WMMA_F16(a0, b2, c[0][2]);
  c[0][3] = WMMA_F16(a0, b3, c[0][3]);
  c[1][0] = WMMA_F16(a1, b0, c[1][0]);
  c[1][1] = WMMA_F16(a1, b1, c[1][1]);
  c[1][2] = WMMA_F16(a1, b2, c[1][2]);
  c[1][3] = WMMA_F16(a1, b3, c[1][3]);

  // D layout: lane L -> col +(L&15); VGPR r -> row + r + (L>>4)*8.
  const int rb = (lane >> 4) << 3;
#pragma unroll
  for (int mi = 0; mi < 2; ++mi) {
#pragma unroll
    for (int ni = 0; ni < 4; ++ni) {
      const int col = n0 + ni * 16 + (lane & 15);
      const float bv = bias[col];
#pragma unroll
      for (int r = 0; r < 8; ++r) {
        size_t idx = (size_t)(m0 + mi * 16 + rb + r) * ldc + col;
        if (F16OUT) ((_Float16*)Cout)[idx] = (_Float16)(c[mi][ni][r] + bv);
        else        ((float*)Cout)[idx]    = c[mi][ni][r] + bv;
      }
    }
  }
}

// ---- RoPE on packed [M, q(2048)|k(2048)] f16 buffer, in place -------------
__global__ void mhsa_rope(_Float16* __restrict__ qk) {
  size_t idx = (size_t)blockIdx.x * blockDim.x + threadIdx.x; // pair index
  if (idx >= (size_t)MROWS * (QKN / 2)) return;
  int m   = (int)(idx >> 11);       // row 0..4095
  int col = ((int)idx & 2047) * 2;  // even element index in row
  int hd  = col & (HDc - 1);        // position of x1 within head dim
  int pos = m & (Sc - 1);           // sequence position
  float ang = (float)pos * __powf(10000.0f, -(float)hd / (float)HDc);
  float sn, cs;
  __sincosf(ang, &sn, &cs);
  _Float16* p = qk + (size_t)m * QKN + col;
  float x1 = (float)p[0], x2 = (float)p[1];
  p[0] = (_Float16)(x1 * cs - x2 * sn);
  p[1] = (_Float16)(x1 * sn + x2 * cs);
}

// ---- fused scores (QK^T / sqrt(H)) + causal mask + softmax ----------------
// Grid: (S/16, NH, B); block 256 (8 waves). Wave w handles k-tile pairs
// (w+16i, w+8+16i): all 8 B-fragments preloaded in one clause, then the
// 8 WMMAs run while later loads are still in flight.
// 16 x 2048 f32 score strip in dynamic LDS (131 KB of the WGP's 320 KB).
#define LDP 2052
__global__ void mhsa_attn(const _Float16* __restrict__ qk,
                          float* __restrict__ out) {
  extern __shared__ float smem[];          // [16][LDP]
  __shared__ float rinv[16];
  const int lane = threadIdx.x & 31;
  const int wave = threadIdx.x >> 5;
  const int qt = blockIdx.x, h = blockIdx.y, b = blockIdx.z;

  const size_t rs = QKN;                   // row stride in halfs
  const _Float16* qbase = qk + ((size_t)b * Sc + qt * 16) * rs + h * HDc;
  const _Float16* kbase = qk + (size_t)b * Sc * rs + Hc + h * HDc;

  const int arow = lane & 15;
  const int aoff = (lane >> 4) << 3;
  const int boff = (lane >> 4) << 3;

  v16h aq[4];
#pragma unroll
  for (int i = 0; i < 4; ++i)
    aq[i] = load_a_frag(qbase + (size_t)arow * rs + i * 32 + aoff);

  const float scale = rsqrtf((float)Hc);   // 1/sqrt(H)
  const int col16 = lane & 15;
  const int rb = (lane >> 4) << 3;

  for (int kt = wave; kt < Sc / 16; kt += 16) {
    const int kt2 = kt + 8;
    v8f c0 = {}, c1 = {};
    const _Float16* pk0 = kbase + (size_t)(kt  * 16 + arow) * rs + boff;
    const _Float16* pk1 = kbase + (size_t)(kt2 * 16 + arow) * rs + boff;

    v16h bf0[4], bf1[4];
#pragma unroll
    for (int i = 0; i < 4; ++i) {
      bf0[i] = load_b_frag(pk0 + i * 32);
      bf1[i] = load_b_frag(pk1 + i * 32);
    }
#pragma unroll
    for (int i = 0; i < 4; ++i) {
      c0 = WMMA_F16(aq[i], bf0[i], c0);
      c1 = WMMA_F16(aq[i], bf1[i], c1);
    }
#pragma unroll
    for (int r = 0; r < 8; ++r) {
      int row = rb + r;
      int sq = qt * 16 + row;
      float v0 = c0[r] * scale;
      float v1 = c1[r] * scale;
      if (kt  * 16 + col16 > sq) v0 = -65504.0f;   // causal mask fill
      if (kt2 * 16 + col16 > sq) v1 = -65504.0f;
      smem[row * LDP + kt  * 16 + col16] = v0;
      smem[row * LDP + kt2 * 16 + col16] = v1;
    }
  }
  __syncthreads();

  // softmax: 16 lanes per row (rows 0..15 <-> thread groups of 16)
  const int row = threadIdx.x >> 4, j = threadIdx.x & 15;
  float* srow = smem + row * LDP;
  float mx = -3.0e38f;
  for (int col = j; col < Sc; col += 16) mx = fmaxf(mx, srow[col]);
#pragma unroll
  for (int s = 1; s < 16; s <<= 1) mx = fmaxf(mx, __shfl_xor(mx, s, 32));
  float sum = 0.0f;
  for (int col = j; col < Sc; col += 16) {
    float e = __expf(srow[col] - mx);
    srow[col] = e;
    sum += e;
  }
#pragma unroll
  for (int s = 1; s < 16; s <<= 1) sum += __shfl_xor(sum, s, 32);
  if (j == 0) rinv[row] = 1.0f / sum;
  __syncthreads();

  // coalesced store of normalized rows
  size_t obase = (((size_t)b * NHc + h) * Sc + qt * 16) * (size_t)Sc;
  for (int idx = threadIdx.x; idx < 16 * Sc; idx += 256) {
    int r = idx >> 11, cc = idx & 2047;
    out[obase + (size_t)r * Sc + cc] = smem[r * LDP + cc] * rinv[r];
  }
}

extern "C" void kernel_launch(void* const* d_in, const int* in_sizes, int n_in,
                              void* d_out, int out_size, void* d_ws, size_t ws_size,
                              hipStream_t stream) {
  (void)in_sizes; (void)n_in; (void)out_size; (void)ws_size;
  const float* x      = (const float*)d_in[0];  // [B,S,H]
  const float* Wqkv_w = (const float*)d_in[1];  // [3H,H] (rows: q|k|v)
  const float* Wqkv_b = (const float*)d_in[2];  // [3H]
  const float* Wo_w   = (const float*)d_in[3];  // [H,H]
  const float* Wo_b   = (const float*)d_in[4];  // [H]
  // d_in[5] = attention_mask (tril) -> implemented as causal mask directly.

  float* xout = (float*)d_out;                               // B*S*H
  float* attn = (float*)d_out + (size_t)Bc * Sc * Hc;        // B*NH*S*S

  // workspace carve-up (all f16, 16B-aligned region sizes)
  _Float16* xh  = (_Float16*)d_ws;                 // [4096,2048]   16 MB
  _Float16* wqk = xh  + (size_t)MROWS * Hc;        // [4096,2048]   16 MB (q|k rows of Wqkv)
  _Float16* woh = wqk + (size_t)QKN * Hc;          // [2048,2048]    8 MB
  _Float16* qkh = woh + (size_t)Hc * Hc;           // [4096,4096]   32 MB

  // 1) converts (vectorized x4)
  mhsa_cvt_f16<<<2048, 256, 0, stream>>>(x,      xh,  (size_t)MROWS * Hc / 4);
  mhsa_cvt_f16<<<2048, 256, 0, stream>>>(Wqkv_w, wqk, (size_t)QKN * Hc / 4);
  mhsa_cvt_f16<<<2048, 256, 0, stream>>>(Wo_w,   woh, (size_t)Hc * Hc / 4);

  // 2) Q,K projection: [4096,2048] x [4096,2048]^T -> f16 [4096,4096]
  mhsa_gemm_wmma<true><<<dim3(QKN / 256, MROWS / 64), 256, 0, stream>>>(
      xh, wqk, Wqkv_b, (void*)qkh, Hc, QKN);

  // 3) RoPE on q and k halves
  mhsa_rope<<<(MROWS * (QKN / 2)) / 256, 256, 0, stream>>>(qkh);

  // 4) fused scores + mask + softmax -> attn
  mhsa_attn<<<dim3(Sc / 16, NHc, Bc), 256, 16 * LDP * sizeof(float), stream>>>(
      qkh, attn);

  // 5) x_out = x @ Wo^T + Wo_b (f32 store)
  mhsa_gemm_wmma<false><<<dim3(Hc / 256, MROWS / 64), 256, 0, stream>>>(
      xh, woh, Wo_b, (void*)xout, Hc, Hc);
}
